// Model_57286273794415
// MI455X (gfx1250) — matrix operand order repaired and checked by reading, not compiled, not run on hardware
//
#include <hip/hip_runtime.h>
#include <hip/hip_bf16.h>

// ---------------------------------------------------------------------------
// Bi-LSTM language model head for MI455X (gfx1250, wave32, WMMA).
//
// All GEMMs use V_WMMA_F32_16X16X32_BF16 (bf16 in, f32 accumulate).
// Dominant cost: final projection [2048 x 1024] x [1024 x 32000] = 134 GFLOP.
// lin_w (65MB bf16) is L2-resident (192MB L2). The projection kernel uses a
// 512M x 64N macro-tile per block: lin_w K-strips are staged in LDS with
// GLOBAL_LOAD_ASYNC_TO_LDS_B128 (inline asm, ASYNCcnt-tracked, double
// buffered) and shared by 8 waves; each wave holds a 64x64 accumulator block
// (16 x v8f = 128 VGPRs). L2 traffic ~2.3GB vs 8.4GB for naive streaming.
// ---------------------------------------------------------------------------

#define V_   32000
#define E_   512
#define H_   512
#define B_   16
#define S_   128
#define G_   2048   // 4*H
#define H2_  1024   // 2*H

typedef __attribute__((ext_vector_type(16))) __bf16 v16bf;
typedef __attribute__((ext_vector_type(8)))  __bf16 v8bf;
typedef __attribute__((ext_vector_type(8)))  float  v8f;

// Async-to-LDS path: device pass only (host pass must not parse the asm with
// x86 constraints). s_wait_asynccnt builtin is probe-confirmed present.
#if defined(__AMDGCN__) && __has_builtin(__builtin_amdgcn_s_wait_asynccnt)
#define USE_ASYNC_LDS 1
#else
#define USE_ASYNC_LDS 0
#endif

// ---- WMMA helpers ----------------------------------------------------------

__device__ __forceinline__ v8f wmma_bf16(v16bf a, v16bf b, v8f c) {
  // D = A(16x32 bf16) x B(32x16 bf16) + C(16x16 f32)
  return __builtin_amdgcn_wmma_f32_16x16x32_bf16(
      /*neg_a=*/false, a, /*neg_b=*/false, b,
      /*c_mod=*/(short)0, c, /*reuse_a=*/false, /*reuse_b=*/false);
}

// A-matrix 16x32 bf16 fragment: lane row = lane&15.
// lanes 0-15 hold K = k0..k0+15; lanes 16-31 hold K = k0+16..k0+31 (matches B).
__device__ __forceinline__ v16bf load_a_frag(const __bf16* row, int k0, int hiK8) {
  return *reinterpret_cast<const v16bf*>(row + k0 + 2 * hiK8);
}

__device__ __forceinline__ float sigmoidf_(float x) {
  return 1.0f / (1.0f + __expf(-x));
}

// ---- elementwise prep ------------------------------------------------------

__global__ void f32_to_bf16_kernel(const float* __restrict__ in,
                                   __bf16* __restrict__ out, long long n) {
  long long i = (long long)blockIdx.x * blockDim.x + threadIdx.x;
  if (i < n) out[i] = (__bf16)in[i];
}

__global__ void bias_combine_kernel(const float* __restrict__ a,
                                    const float* __restrict__ b,
                                    float* __restrict__ o, int n) {
  int i = blockIdx.x * blockDim.x + threadIdx.x;
  if (i < n) o[i] = a[i] + b[i];
}

// x0[s][b][e] = bf16(embed_w[input[b][s]][e])
__global__ void embed_kernel(const int* __restrict__ inp,
                             const float* __restrict__ ew,
                             __bf16* __restrict__ x0) {
  long long idx = (long long)blockIdx.x * blockDim.x + threadIdx.x;
  const long long total = (long long)S_ * B_ * E_;
  if (idx >= total) return;
  int e  = (int)(idx % E_);
  int sb = (int)(idx / E_);
  int b  = sb % B_;
  int s  = sb / B_;
  int tok = inp[b * S_ + s];
  x0[idx] = (__bf16)ew[(long long)tok * E_ + e];
}

// ---- pre-gate GEMM: C[M,N] = A[M,K](bf16) * Bw[N,K]^T(bf16) + bias[N] ------
// One wave computes a 16x64 strip (1 M-tile x 4 N-tiles), A-frag reused 4x.

__global__ __launch_bounds__(256) void gemm_bias_kernel(
    const __bf16* __restrict__ A, const __bf16* __restrict__ Bw,
    const float* __restrict__ bias, float* __restrict__ C,
    int M, int N, int K) {
  const int lane  = threadIdx.x & 31;
  const int wave  = threadIdx.x >> 5;
  const int nBase = (blockIdx.x * 8 + wave) * 64;
  if (nBase >= N) return;                 // wave-uniform exit
  const int mBase = blockIdx.y * 16;
  const int lx    = lane & 15;
  const int hiK16 = (lane >= 16) ? 16 : 0;
  const int hiK8  = (lane >= 16) ? 8  : 0;
  const int hiM8  = (lane >= 16) ? 8  : 0;

  v8f acc[4];
#pragma unroll
  for (int t = 0; t < 4; ++t) {
    float bv = bias[nBase + t * 16 + lx];
#pragma unroll
    for (int r = 0; r < 8; ++r) acc[t][r] = bv;
  }

  const __bf16* arow = A + (size_t)(mBase + lx) * K;
  for (int k0 = 0; k0 < K; k0 += 32) {
    v16bf a = load_a_frag(arow, k0, hiK8);
#pragma unroll
    for (int t = 0; t < 4; ++t) {
      const __bf16* bp = Bw + (size_t)(nBase + t * 16 + lx) * K + k0 + hiK16;
      v16bf bf = *reinterpret_cast<const v16bf*>(bp);
      acc[t] = wmma_bf16(a, bf, acc[t]);
    }
  }

#pragma unroll
  for (int t = 0; t < 4; ++t)
#pragma unroll
    for (int r = 0; r < 8; ++r)
      C[(size_t)(mBase + hiM8 + r) * N + nBase + t * 16 + lx] = acc[t][r];
}

// ---- recurrent LSTM layer --------------------------------------------------
// One 32-wave block per direction. Wave w owns hidden columns [16w,16w+16):
// it computes the i/f/g/o gate tiles for those columns (offsets t*H + j), so
// the cell update is register-local; c is register-resident across all steps;
// h is double-buffered in LDS as bf16 (A operand of next step's WMMA).
// (Cluster barriers would let this span WGPs, but HIP cannot create cluster
// dispatches: ClusterID==0 makes cluster barrier ops NOPs.)

__global__ __launch_bounds__(1024) void lstm_layer_kernel(
    const float* __restrict__ xg_f,  // [S][B][4H] pre-gates, forward dir
    const float* __restrict__ xg_b,  // backward dir
    const __bf16* __restrict__ Whh_f,  // [4H][H] bf16
    const __bf16* __restrict__ Whh_b,
    __bf16* __restrict__ xnext) {      // [S][B][2H] bf16
  __shared__ __bf16 h_lds[2][B_][H_];

  const int dir = blockIdx.x;                    // 0 = forward, 1 = backward
  const float*  xg  = dir ? xg_b  : xg_f;
  const __bf16* Whh = dir ? Whh_b : Whh_f;
  const int dirOff  = dir ? H_ : 0;

  const int lane  = threadIdx.x & 31;
  const int wave  = threadIdx.x >> 5;            // 0..31
  const int j0    = wave * 16;                   // hidden-column base
  const int lx    = lane & 15;
  const int hiK16 = (lane >= 16) ? 16 : 0;
  const int hiK8  = (lane >= 16) ? 8  : 0;
  const int hiM8  = (lane >= 16) ? 8  : 0;

  for (int i = threadIdx.x; i < 2 * B_ * H_; i += 1024)
    (&h_lds[0][0][0])[i] = (__bf16)0.0f;
  __syncthreads();

  v8f c_state;
#pragma unroll
  for (int r = 0; r < 8; ++r) c_state[r] = 0.0f;

  int cur = 0;
  for (int step = 0; step < S_; ++step) {
    const int s = dir ? (S_ - 1 - step) : step;
    const float* xgS = xg + (size_t)s * B_ * G_;

    v8f acc[4];                                   // i, f, g, o tiles
#pragma unroll
    for (int t = 0; t < 4; ++t)
#pragma unroll
      for (int r = 0; r < 8; ++r)
        acc[t][r] = xgS[(size_t)(hiM8 + r) * G_ + t * H_ + j0 + lx];

    const __bf16* hrow = &h_lds[cur][lx][0];
    for (int k0 = 0; k0 < H_; k0 += 32) {
      v16bf a = load_a_frag(hrow, k0, hiK8);
#pragma unroll
      for (int t = 0; t < 4; ++t) {
        const __bf16* bp = Whh + (size_t)(t * H_ + j0 + lx) * H_ + k0 + hiK16;
        v16bf bf = *reinterpret_cast<const v16bf*>(bp);
        acc[t] = wmma_bf16(a, bf, acc[t]);
      }
    }

    const int nxt = cur ^ 1;
#pragma unroll
    for (int r = 0; r < 8; ++r) {
      float ig = sigmoidf_(acc[0][r]);
      float fg = sigmoidf_(acc[1][r]);
      float gg = tanhf(acc[2][r]);
      float og = sigmoidf_(acc[3][r]);
      float cc = fg * c_state[r] + ig * gg;
      c_state[r] = cc;
      float hh = og * tanhf(cc);
      __bf16 hb = (__bf16)hh;
      h_lds[nxt][hiM8 + r][j0 + lx] = hb;
      xnext[((size_t)s * B_ + hiM8 + r) * H2_ + dirOff + j0 + lx] = hb;
    }
    __syncthreads();
    cur = nxt;
  }
}

// ---- final projection: out[b][v][s] = X[s][b][:] . lin_w[v][:] + lin_b[v] --
// Block = 8 waves = 512M x 64N macro-tile; wave = 4 M-tiles x 4 N-tiles.
// lin_w strip (64 rows x 128 K) staged in LDS, double buffered, loaded with
// GLOBAL_LOAD_ASYNC_TO_LDS_B128 (inline asm; VDST = per-lane LDS byte addr,
// VADDR = 64-bit global addr; ASYNCcnt waits), else plain load + ds_store.
// LDS rows padded to 272B so the 32B-per-lane fragment reads stagger 4
// banks/row -> conflict-free b128 ds loads.
// M-rows ordered b*128+s: each lane's 8 acc values hit 8 consecutive s, so
// stores are two 16B vector writes per tile per lane.

#define FP_KC   128                 // K-chunk staged per buffer
#define FP_PAD  136                 // 128 + 8 halfs padding (272B row stride)

__global__ __launch_bounds__(256) void final_proj_kernel(
    const __bf16* __restrict__ X,    // [S][B][2H] bf16
    const __bf16* __restrict__ Wl,   // [V][2H] bf16
    const float* __restrict__ bl,    // [V]
    float* __restrict__ out) {       // [B][V][S]
  __shared__ __bf16 btile[2][64][FP_PAD];

  const int K     = H2_;
  const int lane  = threadIdx.x & 31;
  const int wave  = threadIdx.x >> 5;
  const int nBase = blockIdx.x * 64;              // shared by all 8 waves
  const int mBase = blockIdx.y * 512 + wave * 64; // wave's 64 rows (4 M-tiles)
  const int lx    = lane & 15;
  const int hiK16 = (lane >= 16) ? 16 : 0;
  const int hiK8  = (lane >= 16) ? 8  : 0;
  const int hiM8  = (lane >= 16) ? 8  : 0;

  // stage one 64 x FP_KC strip of Wl into btile[buf]
  auto stage = [&](int buf, int kc) {
#pragma unroll
    for (int i = 0; i < 4; ++i) {
      int g   = threadIdx.x + i * 256;  // 1024 granules of 16B
      int row = g >> 4;
      int cg  = g & 15;
      const __bf16* gp = Wl + (size_t)(nBase + row) * K + kc + cg * 8;
      __bf16*       lp = &btile[buf][row][cg * 8];
#if USE_ASYNC_LDS
      // flat aperture: low 32 bits of a generic LDS pointer == LDS byte addr
      unsigned ldsaddr = (unsigned)(uintptr_t)lp;
      asm volatile("global_load_async_to_lds_b128 %0, %1, off"
                   :
                   : "v"(ldsaddr), "v"(gp)
                   : "memory");
#else
      *reinterpret_cast<float4*>(lp) = *reinterpret_cast<const float4*>(gp);
#endif
    }
  };

  // per-lane A row pointers: m = b*128 + s -> X[s][b][:]
  const __bf16* arow[4];
#pragma unroll
  for (int mt = 0; mt < 4; ++mt) {
    int m = mBase + mt * 16 + lx;
    arow[mt] = X + ((size_t)(m & 127) * B_ + (m >> 7)) * K;
  }

  v8f acc[4][4];  // [mt][nt]
#pragma unroll
  for (int t = 0; t < 4; ++t) {
    float bv = bl[nBase + t * 16 + lx];
#pragma unroll
    for (int mt = 0; mt < 4; ++mt)
#pragma unroll
      for (int r = 0; r < 8; ++r) acc[mt][t][r] = bv;
  }

  const int NCHUNK = K / FP_KC;  // 8
  stage(0, 0);
  for (int c = 0; c < NCHUNK; ++c) {
    const int buf = c & 1;
#if USE_ASYNC_LDS
    __builtin_amdgcn_s_wait_asynccnt(0);   // own chunk-c loads complete
#endif
    __syncthreads();                       // everyone's chunk c visible
    if (c + 1 < NCHUNK) stage(buf ^ 1, (c + 1) * FP_KC);

#pragma unroll
    for (int kk = 0; kk < 4; ++kk) {
      const int k0 = c * FP_KC + kk * 32;
      v16bf a[4];
#pragma unroll
      for (int mt = 0; mt < 4; ++mt) a[mt] = load_a_frag(arow[mt], k0, hiK8);
#pragma unroll
      for (int t = 0; t < 4; ++t) {
        v16bf bf = *reinterpret_cast<const v16bf*>(
            &btile[buf][t * 16 + lx][kk * 32 + hiK16]);
#pragma unroll
        for (int mt = 0; mt < 4; ++mt)
          acc[mt][t] = wmma_bf16(a[mt], bf, acc[mt][t]);
      }
    }
    __syncthreads();                       // reads done before buf is refilled
  }

#pragma unroll
  for (int mt = 0; mt < 4; ++mt) {
    const int mB = mBase + mt * 16;
    const int bT = mB >> 7;
    const int sB = (mB & 127) + hiM8;
#pragma unroll
    for (int t = 0; t < 4; ++t) {
      int v = nBase + t * 16 + lx;
      float* o = out + ((size_t)bT * V_ + v) * S_ + sB;
      float4 lo, hi;
      lo.x = acc[mt][t][0]; lo.y = acc[mt][t][1];
      lo.z = acc[mt][t][2]; lo.w = acc[mt][t][3];
      hi.x = acc[mt][t][4]; hi.y = acc[mt][t][5];
      hi.z = acc[mt][t][6]; hi.w = acc[mt][t][7];
      *reinterpret_cast<float4*>(o)     = lo;
      *reinterpret_cast<float4*>(o + 4) = hi;
    }
  }
}

// ---------------------------------------------------------------------------

extern "C" void kernel_launch(void* const* d_in, const int* in_sizes, int n_in,
                              void* d_out, int out_size, void* d_ws, size_t ws_size,
                              hipStream_t stream) {
  (void)in_sizes; (void)n_in; (void)out_size; (void)ws_size;

  const int*   inp     = (const int*)d_in[0];
  const float* embed_w = (const float*)d_in[1];
  const float* Wih[4]  = { (const float*)d_in[2],  (const float*)d_in[6],
                           (const float*)d_in[10], (const float*)d_in[14] };
  const float* Whh[4]  = { (const float*)d_in[3],  (const float*)d_in[7],
                           (const float*)d_in[11], (const float*)d_in[15] };
  const float* bih[4]  = { (const float*)d_in[4],  (const float*)d_in[8],
                           (const float*)d_in[12], (const float*)d_in[16] };
  const float* bhh[4]  = { (const float*)d_in[5],  (const float*)d_in[9],
                           (const float*)d_in[13], (const float*)d_in[17] };
  const float* lin_w   = (const float*)d_in[18];
  const float* lin_b   = (const float*)d_in[19];
  float* out = (float*)d_out;

  // bump allocator over d_ws (total ~130MB)
  char* ws = (char*)d_ws;
  auto alloc = [&](size_t bytes) -> void* {
    void* p = (void*)ws;
    ws += (bytes + 255) & ~(size_t)255;
    return p;
  };
  __bf16* x0   = (__bf16*)alloc(sizeof(__bf16) * (size_t)S_ * B_ * E_);
  __bf16* x1   = (__bf16*)alloc(sizeof(__bf16) * (size_t)S_ * B_ * H2_);
  __bf16* x2   = (__bf16*)alloc(sizeof(__bf16) * (size_t)S_ * B_ * H2_);
  __bf16* WihC[4]; __bf16* WhhC[4]; float* biasC[4];
  const int wihK[4] = { E_, E_, H2_, H2_ };
  for (int i = 0; i < 4; ++i) {
    WihC[i]  = (__bf16*)alloc(sizeof(__bf16) * (size_t)G_ * wihK[i]);
    WhhC[i]  = (__bf16*)alloc(sizeof(__bf16) * (size_t)G_ * H_);
    biasC[i] = (float*)alloc(sizeof(float) * G_);
  }
  __bf16* linC = (__bf16*)alloc(sizeof(__bf16) * (size_t)V_ * H2_);
  float*  xg_f = (float*)alloc(sizeof(float) * (size_t)S_ * B_ * G_);
  float*  xg_b = (float*)alloc(sizeof(float) * (size_t)S_ * B_ * G_);

  auto cblocks = [](long long n) { return (unsigned)((n + 255) / 256); };

  // 1) convert weights to bf16, fold biases
  for (int i = 0; i < 4; ++i) {
    long long nih = (long long)G_ * wihK[i];
    f32_to_bf16_kernel<<<cblocks(nih), 256, 0, stream>>>(Wih[i], WihC[i], nih);
    long long nhh = (long long)G_ * H_;
    f32_to_bf16_kernel<<<cblocks(nhh), 256, 0, stream>>>(Whh[i], WhhC[i], nhh);
    bias_combine_kernel<<<cblocks(G_), 256, 0, stream>>>(bih[i], bhh[i], biasC[i], G_);
  }
  {
    long long nl = (long long)V_ * H2_;
    f32_to_bf16_kernel<<<cblocks(nl), 256, 0, stream>>>(lin_w, linC, nl);
  }

  // 2) embedding gather -> x0 [S][B][E] bf16
  embed_kernel<<<cblocks((long long)S_ * B_ * E_), 256, 0, stream>>>(inp, embed_w, x0);

  const int M = S_ * B_;  // 2048
  // 3) layer 0: pre-gates then recurrence
  {
    dim3 grid((G_ + 511) / 512, M / 16);
    gemm_bias_kernel<<<grid, 256, 0, stream>>>(x0, WihC[0], biasC[0], xg_f, M, G_, E_);
    gemm_bias_kernel<<<grid, 256, 0, stream>>>(x0, WihC[1], biasC[1], xg_b, M, G_, E_);
    lstm_layer_kernel<<<2, 1024, 0, stream>>>(xg_f, xg_b, WhhC[0], WhhC[1], x1);
  }
  // 4) layer 1
  {
    dim3 grid((G_ + 511) / 512, M / 16);
    gemm_bias_kernel<<<grid, 256, 0, stream>>>(x1, WihC[2], biasC[2], xg_f, M, G_, H2_);
    gemm_bias_kernel<<<grid, 256, 0, stream>>>(x1, WihC[3], biasC[3], xg_b, M, G_, H2_);
    lstm_layer_kernel<<<2, 1024, 0, stream>>>(xg_f, xg_b, WhhC[2], WhhC[3], x2);
  }
  // 5) final projection -> out [B][V][S], 512M x 64N macro-tiles
  {
    dim3 grid(V_ / 64, M / 512);   // (500, 4)
    final_proj_kernel<<<grid, 256, 0, stream>>>(x2, linC, lin_b, out);
  }
}
